// SV_DGCNN_CLS_63007170232737
// MI455X (gfx1250) — matrix operand order repaired and checked by reading, not compiled, not run on hardware
//
#include <hip/hip_runtime.h>

#define NBATCH 32
#define NPTS   1024
#define KNN_K  20
#define SLOPE  0.2f
#define EPSF   1e-5f

typedef __attribute__((ext_vector_type(16))) _Float16 v16h;
typedef __attribute__((ext_vector_type(2)))  _Float16 v2h;
typedef __attribute__((ext_vector_type(8)))  float    v8f;

// =====================================================================
// WMMA GEMM: OUT(O,M) = W(O,C) * X(C,M).  f32 in/out, f16 compute via
// v_wmma_f32_16x16x32_f16.  Block = 128 threads (4 waves), 32x32 tile,
// 64-deep K slab per barrier (2 WMMAs back-to-back per wave).
// LDS tiles are stored in WMMA *fragment order* so each lane fetches its
// whole v16h fragment with contiguous ds_load_b128s (no scalar gathers).
// =====================================================================
__global__ __launch_bounds__(128) void gemm_wmma_kernel(
    const float* __restrict__ W, const float* __restrict__ X,
    float* __restrict__ OUT, int O, int C, int M) {
  __shared__ v16h AF[2][2][32];   // [ksub][osub][lane]
  __shared__ v16h BF[2][2][32];   // [ksub][msub][lane]
  _Float16* afp = (_Float16*)AF;
  _Float16* bfp = (_Float16*)BF;
  const int tid  = threadIdx.x;
  const int wave = tid >> 5;
  const int lane = tid & 31;
  const int o0 = blockIdx.y * 32;
  const int m0 = blockIdx.x * 32;
  const int osub = wave >> 1;      // which 16-row block of the 32x32 tile
  const int msub = wave & 1;       // which 16-col block
  const int half = lane >> 4;
  const int mrow = lane & 15;
  v8f acc = {0.f, 0.f, 0.f, 0.f, 0.f, 0.f, 0.f, 0.f};

  for (int k0 = 0; k0 < C; k0 += 64) {
    // ---- stage A (32 o-rows x 64 k), packed f16 pair writes, fragment order
    // A layout (16-bit 16x32): lane half selects K+8 (r<4) / K+24 (r>=4):
    //   k in 32-slab -> hf=(k>>3)&1, e=(k&7)+8*(k>>4), dest lane = hf*16+row
    for (int i = tid; i < 1024; i += 128) {
      int r = i >> 5;              // o row 0..31
      int c = (i & 31) << 1;       // k 0..62 (even)
      int oo = o0 + r, kk = k0 + c;
      float f0 = (oo < O && kk     < C) ? W[(long)oo * C + kk]     : 0.f;
      float f1 = (oo < O && kk + 1 < C) ? W[(long)oo * C + kk + 1] : 0.f;
      int ks = c >> 5, cc = c & 31;
      int hf = (cc >> 3) & 1;
      int e  = (cc & 7) + ((cc >> 4) << 3);
      int dl = (hf << 4) | (r & 15);
      v2h* dst = (v2h*)(afp + ((((ks << 1) | (r >> 4)) * 32 + dl) * 16 + e));
      *dst = (v2h){(_Float16)f0, (_Float16)f1};
    }
    // ---- stage B (64 k x 32 m), fragment order
    // B layout (32x16): dest lane = ((k>>3)&1)*16 + (m&15), e = (k&7)+8*(k>>4)
    for (int i = tid; i < 2048; i += 128) {
      int r = i >> 5;              // k 0..63
      int c = i & 31;              // m 0..31 (coalesced on global X)
      int kk = k0 + r, mm = m0 + c;
      float f = (kk < C && mm < M) ? X[(long)kk * M + mm] : 0.f;
      int ks = r >> 5, rr = r & 31;
      int dl = (((rr >> 3) & 1) << 4) | (c & 15);
      bfp[(((ks << 1) | (c >> 4)) * 32 + dl) * 16 + ((rr & 7) + ((rr >> 4) << 3))] = (_Float16)f;
    }
    // prefetch next K slab (global_prefetch_b8 on gfx1250)
    if (k0 + 64 < C) {
      int ro = o0 + lane;
      if (ro < O) __builtin_prefetch(W + (long)ro * C + (k0 + 64), 0, 1);
      int rk = k0 + 64 + wave, cm = m0 + lane;
      if (rk < C && cm < M) __builtin_prefetch(X + (long)rk * M + cm, 0, 1);
    }
    __syncthreads();

    v16h a0 = AF[0][osub][lane];
    v16h b0 = BF[0][msub][lane];
    acc = __builtin_amdgcn_wmma_f32_16x16x32_f16(
        false, a0, false, b0, (short)0, acc, false, false);
    v16h a1 = AF[1][osub][lane];
    v16h b1 = BF[1][msub][lane];
    acc = __builtin_amdgcn_wmma_f32_16x16x32_f16(
        false, a1, false, b1, (short)0, acc, false, false);
    __syncthreads();
  }
  // C/D layout: lane -> column N, vgpr j -> row M = j + 8*half
#pragma unroll
  for (int j = 0; j < 8; ++j) {
    int oo = o0 + (osub << 4) + (half << 3) + j;
    int mm = m0 + (msub << 4) + mrow;
    if (oo < O && mm < M) OUT[(long)oo * M + mm] = acc[j];
  }
}

// =====================================================================
// xx[b][i] = sum_c f[b][c][i]^2
// =====================================================================
__global__ void xx_kernel(const float* __restrict__ F, float* __restrict__ XX,
                          int C, int N) {
  int b = blockIdx.y;
  int i = blockIdx.x * blockDim.x + threadIdx.x;
  if (i >= N) return;
  const float* Fb = F + (long)b * C * N;
  float s = 0.f;
  for (int c = 0; c < C; ++c) { float v = Fb[c * N + i]; s += v * v; }
  XX[b * N + i] = s;
}

// =====================================================================
// KNN: one block per (b,i).  neg_d = 2*<fi,fj> - xx_i - xx_j, extract
// top-KNN_K by iterative block argmax over LDS-resident distances.
// =====================================================================
__global__ __launch_bounds__(256) void knn_kernel(
    const float* __restrict__ F, const float* __restrict__ XX,
    int* __restrict__ IDX, int C, int N, int Ktop) {
  __shared__ float fi[128];
  __shared__ float dist[NPTS];
  __shared__ float rval[256];
  __shared__ int   ridx[256];
  int b = blockIdx.y;
  int i = blockIdx.x;
  const float* Fb = F + (long)b * C * N;
  for (int c = threadIdx.x; c < C; c += blockDim.x) fi[c] = Fb[c * N + i];
  __syncthreads();
  float xxi = XX[b * N + i];
  for (int j = threadIdx.x; j < N; j += blockDim.x) {
    float dot = 0.f;
    for (int c = 0; c < C; ++c) dot += fi[c] * Fb[c * N + j];
    dist[j] = 2.f * dot - xxi - XX[b * N + j];
  }
  __syncthreads();
  int* idxo = IDX + ((long)b * N + i) * Ktop;
  for (int t = 0; t < Ktop; ++t) {
    float best = -3.4e38f; int bi = 0;
    for (int j = threadIdx.x; j < N; j += blockDim.x) {
      float v = dist[j];
      if (v > best) { best = v; bi = j; }
    }
    rval[threadIdx.x] = best; ridx[threadIdx.x] = bi;
    __syncthreads();
    for (int s = 128; s > 0; s >>= 1) {
      if ((int)threadIdx.x < s) {
        if (rval[threadIdx.x + s] > rval[threadIdx.x]) {
          rval[threadIdx.x] = rval[threadIdx.x + s];
          ridx[threadIdx.x] = ridx[threadIdx.x + s];
        }
      }
      __syncthreads();
    }
    if (threadIdx.x == 0) { idxo[t] = ridx[0]; dist[ridx[0]] = -3.4e38f; }
    __syncthreads();
  }
}

// =====================================================================
// pack F(b, sp + 3*vp, n) = [ s_pool ; v_pool.reshape(vp*3) ]
// =====================================================================
__global__ void pack_feat_kernel(const float* __restrict__ SP,
                                 const float* __restrict__ VP,
                                 float* __restrict__ F, int sp, int vp, int N,
                                 int spool_ch, int vpool_ch, int soff, int voff) {
  int b = blockIdx.y;
  int n = blockIdx.x * blockDim.x + threadIdx.x;
  if (n >= N) return;
  int C = sp + 3 * vp;
  float* Fb = F + (long)b * C * N;
  const float* Sb = SP + ((long)b * spool_ch + soff) * N;
  const float* Vb = VP + ((long)b * vpool_ch + voff) * 3 * N;
  for (int c = 0; c < sp; ++c) Fb[c * N + n] = Sb[c * N + n];
  for (int cd = 0; cd < 3 * vp; ++cd) Fb[(sp + cd) * N + n] = Vb[cd * N + n];
}

// =====================================================================
// Stage-1 edge features (per batch b): v=[xj-xi, xi], Vector2Scalar
// invariants, and vector norms.  SC: (8, N*K), VE: (2,3,N*K)
// =====================================================================
__global__ void edge1_kernel(const float* __restrict__ X, const int* __restrict__ IDX,
                             const float* __restrict__ W2, float* __restrict__ SC,
                             float* __restrict__ VE, int b, int N, int Ktop) {
  int NK = N * Ktop;
  int m = blockIdx.x * blockDim.x + threadIdx.x;
  if (m >= NK) return;
  int n = m / Ktop, k = m - n * Ktop;
  int j = IDX[((long)b * N + n) * Ktop + k];
  const float* Xb = X + (long)b * 3 * N;
  float e[2][3];
  for (int d = 0; d < 3; ++d) {
    float xi = Xb[d * N + n], xj = Xb[d * N + j];
    e[0][d] = xj - xi; e[1][d] = xi;
  }
  for (int c = 0; c < 2; ++c)
    for (int d = 0; d < 3; ++d) VE[(long)(c * 3 + d) * NK + m] = e[c][d];
  float z[3][3];
  for (int mm = 0; mm < 3; ++mm)
    for (int d = 0; d < 3; ++d)
      z[mm][d] = W2[mm * 2 + 0] * e[0][d] + W2[mm * 2 + 1] * e[1][d];
  for (int c = 0; c < 2; ++c)
    for (int mm = 0; mm < 3; ++mm) {
      float s = 0.f;
      for (int d = 0; d < 3; ++d) s += e[c][d] * z[mm][d];
      SC[(long)(c * 3 + mm) * NK + m] = s;
    }
  for (int c = 0; c < 2; ++c) {
    float s = 0.f;
    for (int d = 0; d < 3; ++d) s += e[c][d] * e[c][d];
    SC[(long)(6 + c) * NK + m] = sqrtf(s);
  }
}

// =====================================================================
// Generic edge features (stages 2-4, per batch b):
// SC = [sj-si (sp) ; si (sp) ; ||vj-vi|| (vp) ; ||vi|| (vp)]
// VE = [vj-vi (vp,3) ; vi (vp,3)]
// =====================================================================
__global__ void edge_kernel(const float* __restrict__ SP, const float* __restrict__ VP,
                            const int* __restrict__ IDX, float* __restrict__ SC,
                            float* __restrict__ VE, int b, int sp, int vp,
                            int N, int Ktop, int spool_ch, int vpool_ch,
                            int soff, int voff) {
  int NK = N * Ktop;
  int m = blockIdx.x * blockDim.x + threadIdx.x;
  if (m >= NK) return;
  int n = m / Ktop, k = m - n * Ktop;
  int j = IDX[((long)b * N + n) * Ktop + k];
  const float* Sb = SP + ((long)b * spool_ch + soff) * N;
  const float* Vb = VP + ((long)b * vpool_ch + voff) * 3 * N;
  for (int c = 0; c < sp; ++c) {
    float si = Sb[c * N + n], sj = Sb[c * N + j];
    SC[(long)c * NK + m] = sj - si;
    SC[(long)(sp + c) * NK + m] = si;
  }
  for (int c = 0; c < vp; ++c) {
    float n0 = 0.f, n1 = 0.f;
    for (int d = 0; d < 3; ++d) {
      float vi = Vb[(c * 3 + d) * N + n], vj = Vb[(c * 3 + d) * N + j];
      float df = vj - vi;
      VE[(long)(c * 3 + d) * NK + m] = df;
      VE[(long)((vp + c) * 3 + d) * NK + m] = vi;
      n0 += df * df; n1 += vi * vi;
    }
    SC[(long)(2 * sp + c) * NK + m]      = sqrtf(n0);
    SC[(long)(2 * sp + vp + c) * NK + m] = sqrtf(n1);
  }
}

// =====================================================================
// VN-style equivariant leaky relu, in place on V1:
// v_out = v1 - (1-slope)*min(dot(v1,d),0)/(||d||^2+eps) * d
// layout (vo, 3, M)
// =====================================================================
__global__ void vnrelu_kernel(float* __restrict__ V1, const float* __restrict__ D,
                              int vo, long M) {
  long t = (long)blockIdx.x * blockDim.x + threadIdx.x;
  if (t >= (long)vo * M) return;
  int c = (int)(t / M);
  long m = t - (long)c * M;
  long b0 = (long)c * 3 * M + m;
  float v0 = V1[b0], v1 = V1[b0 + M], v2 = V1[b0 + 2 * M];
  float d0 = D[b0],  d1 = D[b0 + M],  d2 = D[b0 + 2 * M];
  float dot = v0 * d0 + v1 * d1 + v2 * d2;
  float dsq = d0 * d0 + d1 * d1 + d2 * d2 + EPSF;
  float coef = dot < 0.f ? (1.f - SLOPE) * dot / dsq : 0.f;
  V1[b0]         = v0 - coef * d0;
  V1[b0 + M]     = v1 - coef * d1;
  V1[b0 + 2 * M] = v2 - coef * d2;
}

// =====================================================================
// vector pool: mean over K.  V1 (vo,3,N*K) -> VPOOL[b] slice
// =====================================================================
__global__ void vpool_kernel(const float* __restrict__ V1, float* __restrict__ VP,
                             int b, int vo, int N, int Ktop, int vpool_ch, int voff) {
  int t = blockIdx.x * blockDim.x + threadIdx.x;
  if (t >= vo * 3 * N) return;
  int cd = t / N, n = t - cd * N;
  const float* src = V1 + (long)cd * N * Ktop + (long)n * Ktop;
  float s = 0.f;
  for (int k = 0; k < Ktop; ++k) s += src[k];
  VP[((long)b * vpool_ch * 3 + voff * 3 + cd) * N + n] = s / Ktop;
}

// =====================================================================
// scalar pool: max over K.  S (B,so,N*K) -> SPOOL slice
// =====================================================================
__global__ void spool_kernel(const float* __restrict__ S, float* __restrict__ SP,
                             int so, int N, int Ktop, int Bn, int spool_ch, int soff) {
  long t = (long)blockIdx.x * blockDim.x + threadIdx.x;
  if (t >= (long)Bn * so * N) return;
  int n = (int)(t % N);
  int c = (int)((t / N) % so);
  int b = (int)(t / ((long)N * so));
  const float* src = S + (((long)b * so + c) * N + n) * Ktop;
  float mx = src[0];
  for (int k = 1; k < Ktop; ++k) mx = fmaxf(mx, src[k]);
  SP[((long)b * spool_ch + soff + c) * N + n] = mx;
}

// =====================================================================
// BatchNorm stats: one block per channel over (B, M) elements
// =====================================================================
__global__ __launch_bounds__(256) void bnstat_kernel(
    const float* __restrict__ S, float* __restrict__ MV, int Bn, int so, long M) {
  __shared__ float rs[256], rs2[256];
  int c = blockIdx.x;
  float s = 0.f, s2 = 0.f;
  for (int b = 0; b < Bn; ++b) {
    const float* p = S + ((long)b * so + c) * M;
    for (long m = threadIdx.x; m < M; m += blockDim.x) {
      float v = p[m]; s += v; s2 += v * v;
    }
  }
  rs[threadIdx.x] = s; rs2[threadIdx.x] = s2;
  __syncthreads();
  for (int st = 128; st > 0; st >>= 1) {
    if ((int)threadIdx.x < st) {
      rs[threadIdx.x] += rs[threadIdx.x + st];
      rs2[threadIdx.x] += rs2[threadIdx.x + st];
    }
    __syncthreads();
  }
  if (threadIdx.x == 0) {
    float cnt = (float)((long)Bn * M);
    float mu = rs[0] / cnt;
    MV[2 * c] = mu;
    MV[2 * c + 1] = rs2[0] / cnt - mu * mu;
  }
}

// =====================================================================
// BN apply + leaky relu, in place.  S layout (B, so, M) contiguous.
// =====================================================================
__global__ void bnapply_kernel(float* __restrict__ S, const float* __restrict__ MV,
                               const float* __restrict__ G, const float* __restrict__ Bi,
                               int Bn, int so, long M) {
  long t = (long)blockIdx.x * blockDim.x + threadIdx.x;
  if (t >= (long)Bn * so * M) return;
  int c = (int)((t / M) % so);
  float mu = MV[2 * c], var = MV[2 * c + 1];
  float y = G[c] * (S[t] - mu) * rsqrtf(var + EPSF) + Bi[c];
  S[t] = y > 0.f ? y : SLOPE * y;
}

// =====================================================================
// Stage-5 scalar input: SC(339,N) = [spool_cat(256) ; ||vpool_cat||(83)]
// =====================================================================
__global__ void sc5_kernel(const float* __restrict__ SP, const float* __restrict__ VP,
                           float* __restrict__ SC, int b, int N) {
  int n = blockIdx.x * blockDim.x + threadIdx.x;
  if (n >= N) return;
  const float* Sb = SP + (long)b * 256 * N;
  const float* Vb = VP + (long)b * 83 * 3 * N;
  for (int c = 0; c < 256; ++c) SC[(long)c * N + n] = Sb[c * N + n];
  for (int c = 0; c < 83; ++c) {
    float s = 0.f;
    for (int d = 0; d < 3; ++d) { float v = Vb[(c * 3 + d) * N + n]; s += v * v; }
    SC[(long)(256 + c) * N + n] = sqrtf(s);
  }
}

// =====================================================================
// SVFuse: zf[m][d] = sum_c FW[m][c]*v[c][d];  sv[c*3+m] = sum_d v[c][d]*zf[m][d]
// =====================================================================
__global__ void fuse_kernel(const float* __restrict__ V, const float* __restrict__ FW,
                            float* __restrict__ SV, int b, int N) {
  int n = blockIdx.x * blockDim.x + threadIdx.x;
  if (n >= N) return;
  const float* Vb = V + (long)b * 170 * 3 * N;
  float* SVb = SV + (long)b * 510 * N;
  float z[3][3];
  for (int mm = 0; mm < 3; ++mm)
    for (int d = 0; d < 3; ++d) {
      float s = 0.f;
      for (int c = 0; c < 170; ++c) s += FW[mm * 170 + c] * Vb[(c * 3 + d) * N + n];
      z[mm][d] = s;
    }
  for (int c = 0; c < 170; ++c) {
    float v0 = Vb[(c * 3 + 0) * N + n];
    float v1 = Vb[(c * 3 + 1) * N + n];
    float v2 = Vb[(c * 3 + 2) * N + n];
    for (int mm = 0; mm < 3; ++mm)
      SVb[(long)(c * 3 + mm) * N + n] = v0 * z[mm][0] + v1 * z[mm][1] + v2 * z[mm][2];
  }
}

// =====================================================================
// Global pool over N: writes HT transposed (2044 x B):
// rows [0,1022) = max, rows [1022,2044) = mean
// =====================================================================
__global__ __launch_bounds__(256) void hpool_kernel(
    const float* __restrict__ S5, const float* __restrict__ SV,
    float* __restrict__ HT, int N, int Bn) {
  __shared__ float smx[256], ssm[256];
  int ch = blockIdx.x, b = blockIdx.y;
  const float* src = (ch < 512) ? (S5 + ((long)b * 512 + ch) * N)
                                : (SV + ((long)b * 510 + (ch - 512)) * N);
  float mx = -3.4e38f, sm = 0.f;
  for (int n = threadIdx.x; n < N; n += blockDim.x) {
    float v = src[n]; mx = fmaxf(mx, v); sm += v;
  }
  smx[threadIdx.x] = mx; ssm[threadIdx.x] = sm;
  __syncthreads();
  for (int s = 128; s > 0; s >>= 1) {
    if ((int)threadIdx.x < s) {
      smx[threadIdx.x] = fmaxf(smx[threadIdx.x], smx[threadIdx.x + s]);
      ssm[threadIdx.x] += ssm[threadIdx.x + s];
    }
    __syncthreads();
  }
  if (threadIdx.x == 0) {
    HT[(long)ch * Bn + b] = smx[0];
    HT[(long)(1022 + ch) * Bn + b] = ssm[0] / (float)N;
  }
}

// =====================================================================
// Head BN (over batch dim) + leaky, in place.  H layout (O, Bn)
// =====================================================================
__global__ void bnhead_kernel(float* __restrict__ H, const float* __restrict__ G,
                              const float* __restrict__ Bi, int O, int Bn) {
  int c = blockIdx.x * blockDim.x + threadIdx.x;
  if (c >= O) return;
  float* row = H + (long)c * Bn;
  float s = 0.f, s2 = 0.f;
  for (int b = 0; b < Bn; ++b) { float v = row[b]; s += v; s2 += v * v; }
  float mu = s / Bn, var = s2 / Bn - mu * mu;
  float inv = rsqrtf(var + EPSF);
  for (int b = 0; b < Bn; ++b) {
    float y = G[c] * (row[b] - mu) * inv + Bi[c];
    row[b] = y > 0.f ? y : SLOPE * y;
  }
}

// =====================================================================
// Final: transpose (O,Bn) -> (Bn,O) with bias add
// =====================================================================
__global__ void out_kernel(const float* __restrict__ H3, const float* __restrict__ bias,
                           float* __restrict__ OUT, int O, int Bn) {
  int t = blockIdx.x * blockDim.x + threadIdx.x;
  if (t >= O * Bn) return;
  int b = t / O, o = t - b * O;
  OUT[t] = H3[(long)o * Bn + b] + bias[o];
}

// =====================================================================
extern "C" void kernel_launch(void* const* d_in, const int* in_sizes, int n_in,
                              void* d_out, int out_size, void* d_ws, size_t ws_size,
                              hipStream_t stream) {
  (void)in_sizes; (void)n_in; (void)out_size; (void)ws_size;
  const int B = NBATCH, N = NPTS, K = KNN_K, NK = N * K;

  const float* x     = (const float*)d_in[0];
  const float* v2s_w = (const float*)d_in[1];
  const float *ws[6], *bng[6], *bnb[6], *wv[6], *wd[6];
  for (int i = 1; i <= 5; ++i) {
    int base = 2 + (i - 1) * 5;
    ws[i]  = (const float*)d_in[base];
    bng[i] = (const float*)d_in[base + 1];
    bnb[i] = (const float*)d_in[base + 2];
    wv[i]  = (const float*)d_in[base + 3];
    wd[i]  = (const float*)d_in[base + 4];
  }
  const float* fuse_w = (const float*)d_in[27];
  const float* lin1_w = (const float*)d_in[28];
  const float* bn1_g  = (const float*)d_in[29];
  const float* bn1_b  = (const float*)d_in[30];
  const float* lin2_w = (const float*)d_in[31];
  const float* bn2_g  = (const float*)d_in[32];
  const float* bn2_b  = (const float*)d_in[33];
  const float* lin3_w = (const float*)d_in[34];
  const float* lin3_b = (const float*)d_in[35];

  // ---- workspace carve-up (f32 elements) ----
  char* base8 = (char*)d_ws;
  size_t off = 0;
  auto take = [&](size_t nelem) -> float* {
    float* p = (float*)(base8 + off);
    off += (nelem * 4 + 255) & ~(size_t)255;
    return p;
  };
  float* F     = take((size_t)B * 128 * N);
  float* XXb   = take((size_t)B * N);
  int*   IDX   = (int*)take((size_t)B * N * K);
  float* SPRE  = take((size_t)B * 128 * NK);      // pre-BN scalar activations
  float* SPOOL = take((size_t)B * 256 * N);       // s1..s4 pooled (ch offsets 0/32/64/128)
  float* VPOOL = take((size_t)B * 83 * 3 * N);    // v1..v4 pooled (ch offsets 0/10/20/41)
  float* V5    = take((size_t)B * 170 * 3 * N);
  float* SV    = take((size_t)B * 510 * N);
  float* HT    = take((size_t)2044 * B);
  float* H1    = take((size_t)512 * B);
  float* H2    = take((size_t)256 * B);
  float* H3    = take((size_t)48 * B);
  float* MV    = take((size_t)2 * 512);
  float* SCB   = take((size_t)339 * NK);          // per-batch scalar GEMM input
  float* VEB   = take((size_t)84 * 3 * NK);       // per-batch vector edge features
  float* V1B   = take((size_t)84 * 3 * NK);       // per-batch wv output
  float* DB    = take((size_t)84 * 3 * NK);       // per-batch wd output

  auto gemm = [&](const float* Wp, const float* Xp, float* Op, int O, int C, int M) {
    dim3 g((unsigned)((M + 31) / 32), (unsigned)((O + 31) / 32));
    gemm_wmma_kernel<<<g, 128, 0, stream>>>(Wp, Xp, Op, O, C, M);
  };

  // ================= Stage 1 (coords) =================
  xx_kernel<<<dim3((N + 255) / 256, B), 256, 0, stream>>>(x, XXb, 3, N);
  knn_kernel<<<dim3(N, B), 256, 0, stream>>>(x, XXb, IDX, 3, N, K);
  for (int b = 0; b < B; ++b) {
    edge1_kernel<<<(NK + 255) / 256, 256, 0, stream>>>(x, IDX, v2s_w, SCB, VEB, b, N, K);
    gemm(ws[1], SCB, SPRE + (size_t)b * 32 * NK, 32, 8, NK);
    gemm(wv[1], VEB, V1B, 10, 2, 3 * NK);
    gemm(wd[1], VEB, DB, 10, 2, 3 * NK);
    vnrelu_kernel<<<(unsigned)(((long)10 * NK + 255) / 256), 256, 0, stream>>>(V1B, DB, 10, (long)NK);
    vpool_kernel<<<(10 * 3 * N + 255) / 256, 256, 0, stream>>>(V1B, VPOOL, b, 10, N, K, 83, 0);
  }
  bnstat_kernel<<<32, 256, 0, stream>>>(SPRE, MV, B, 32, (long)NK);
  bnapply_kernel<<<(unsigned)(((long)B * 32 * NK + 255) / 256), 256, 0, stream>>>(
      SPRE, MV, bng[1], bnb[1], B, 32, (long)NK);
  spool_kernel<<<(unsigned)(((long)B * 32 * N + 255) / 256), 256, 0, stream>>>(
      SPRE, SPOOL, 32, N, K, B, 256, 0);

  // ================= Stages 2-4 =================
  const int sp_in[3]   = {32, 32, 64};
  const int vp_in[3]   = {10, 10, 21};
  const int soff_in[3] = {0, 32, 64};
  const int voff_in[3] = {0, 10, 20};
  const int so_out[3]  = {32, 64, 128};
  const int vo_out[3]  = {10, 21, 42};
  const int soff_o[3]  = {32, 64, 128};
  const int voff_o[3]  = {10, 20, 41};
  for (int s = 0; s < 3; ++s) {
    int widx = s + 2;
    int sp = sp_in[s], vp = vp_in[s];
    int Cf = sp + 3 * vp;
    int so = so_out[s], vo = vo_out[s];
    int Csc = 2 * sp + 2 * vp, Cv = 2 * vp;
    pack_feat_kernel<<<dim3((N + 255) / 256, B), 256, 0, stream>>>(
        SPOOL, VPOOL, F, sp, vp, N, 256, 83, soff_in[s], voff_in[s]);
    xx_kernel<<<dim3((N + 255) / 256, B), 256, 0, stream>>>(F, XXb, Cf, N);
    knn_kernel<<<dim3(N, B), 256, 0, stream>>>(F, XXb, IDX, Cf, N, K);
    for (int b = 0; b < B; ++b) {
      edge_kernel<<<(NK + 255) / 256, 256, 0, stream>>>(
          SPOOL, VPOOL, IDX, SCB, VEB, b, sp, vp, N, K, 256, 83, soff_in[s], voff_in[s]);
      gemm(ws[widx], SCB, SPRE + (size_t)b * so * NK, so, Csc, NK);
      gemm(wv[widx], VEB, V1B, vo, Cv, 3 * NK);
      gemm(wd[widx], VEB, DB, vo, Cv, 3 * NK);
      vnrelu_kernel<<<(unsigned)(((long)vo * NK + 255) / 256), 256, 0, stream>>>(
          V1B, DB, vo, (long)NK);
      vpool_kernel<<<(vo * 3 * N + 255) / 256, 256, 0, stream>>>(
          V1B, VPOOL, b, vo, N, K, 83, voff_o[s]);
    }
    bnstat_kernel<<<so, 256, 0, stream>>>(SPRE, MV, B, so, (long)NK);
    bnapply_kernel<<<(unsigned)(((long)B * so * NK + 255) / 256), 256, 0, stream>>>(
        SPRE, MV, bng[widx], bnb[widx], B, so, (long)NK);
    spool_kernel<<<(unsigned)(((long)B * so * N + 255) / 256), 256, 0, stream>>>(
        SPRE, SPOOL, so, N, K, B, 256, soff_o[s]);
  }

  // ================= Stage 5 (svcat) =================
  for (int b = 0; b < B; ++b) {
    sc5_kernel<<<(N + 255) / 256, 256, 0, stream>>>(SPOOL, VPOOL, SCB, b, N);
    gemm(ws[5], SCB, SPRE + (size_t)b * 512 * N, 512, 339, N);
    const float* vb = VPOOL + (size_t)b * 83 * 3 * N;
    gemm(wv[5], vb, V5 + (size_t)b * 170 * 3 * N, 170, 83, 3 * N);
    gemm(wd[5], vb, DB, 170, 83, 3 * N);
    vnrelu_kernel<<<(unsigned)(((long)170 * N + 255) / 256), 256, 0, stream>>>(
        V5 + (size_t)b * 170 * 3 * N, DB, 170, (long)N);
  }
  bnstat_kernel<<<512, 256, 0, stream>>>(SPRE, MV, B, 512, (long)N);
  bnapply_kernel<<<(unsigned)(((long)B * 512 * N + 255) / 256), 256, 0, stream>>>(
      SPRE, MV, bng[5], bnb[5], B, 512, (long)N);
  for (int b = 0; b < B; ++b)
    fuse_kernel<<<(N + 255) / 256, 256, 0, stream>>>(V5, fuse_w, SV, b, N);
  hpool_kernel<<<dim3(1022, B), 256, 0, stream>>>(SPRE, SV, HT, N, B);

  // ================= Head =================
  gemm(lin1_w, HT, H1, 512, 2044, B);
  bnhead_kernel<<<(512 + 63) / 64, 64, 0, stream>>>(H1, bn1_g, bn1_b, 512, B);
  gemm(lin2_w, H1, H2, 256, 512, B);
  bnhead_kernel<<<(256 + 63) / 64, 64, 0, stream>>>(H2, bn2_g, bn2_b, 256, B);
  gemm(lin3_w, H2, H3, 40, 256, B);
  out_kernel<<<(40 * B + 255) / 256, 256, 0, stream>>>(H3, lin3_b, (float*)d_out, 40, B);
}
